// GCN_7387343749683
// MI455X (gfx1250) — matrix operand order repaired and checked by reading, not compiled, not run on hardware
//
#include <hip/hip_runtime.h>
#include <hip/hip_bf16.h>

typedef __attribute__((ext_vector_type(16))) _Float16 v16h;
typedef __attribute__((ext_vector_type(8)))  _Float16 v8h;
typedef __attribute__((ext_vector_type(8)))  float    v8f;

#define N_NODES 32768
#define G_GR    64
#define NPG_    512
#define F_IN    768
#define H_      256
#define E_EDGES 1048576
#define EN_     (E_EDGES + N_NODES)   // edges + self loops = 1081344

// ---------- order-preserving float <-> uint encode for atomicMax ----------
__device__ __forceinline__ unsigned encf(float f) {
    unsigned u = __float_as_uint(f);
    return (u & 0x80000000u) ? ~u : (u | 0x80000000u);
}
__device__ __forceinline__ float decf(unsigned e) {
    unsigned u = (e & 0x80000000u) ? (e ^ 0x80000000u) : ~e;
    return __uint_as_float(u);
}

// ---------------------------------------------------------------------------
// Kernel 0: split [Wl|Wr] (f32, [768][256] each, col n of fused N=512) into
// transposed hi/lo f16 matrices laid out [512][768] so that a WMMA B fragment
// (two runs of 8 K values for one column) is two contiguous 16B loads.
// ---------------------------------------------------------------------------
__global__ __launch_bounds__(256)
void gat_split_w(const float* __restrict__ Wl, const float* __restrict__ Wr,
                 _Float16* __restrict__ Bhi, _Float16* __restrict__ Blo)
{
    int i = blockIdx.x * 256 + threadIdx.x;          // 512*768 elements
    if (i >= 512 * F_IN) return;
    int n = i / F_IN, k = i - n * F_IN;
    float w = (n < H_) ? Wl[(size_t)k * H_ + n] : Wr[(size_t)k * H_ + (n - H_)];
    _Float16 hi = (_Float16)w;
    Bhi[i] = hi;
    Blo[i] = (_Float16)(w - (float)hi);
}

// ---------------------------------------------------------------------------
// Kernel 1: [xl|xr] = x @ [Wl|Wr] + [bl|br] via split-f16 WMMA (f32-accurate).
// Block = 256 threads (8 waves). Block -> one M=16 tile; wave w -> N-strip
// [w*64, w*64+64). K loop: 24 steps of 32. 3 WMMA per (ktile,ntile).
// ---------------------------------------------------------------------------
__global__ __launch_bounds__(256)
void gat_gemm_xlxr(const float* __restrict__ x,
                   const _Float16* __restrict__ Bhi,
                   const _Float16* __restrict__ Blo,
                   const float* __restrict__ bl, const float* __restrict__ br,
                   float* __restrict__ xl, float* __restrict__ xr)
{
    __shared__ _Float16 Ahi[16][32];
    __shared__ _Float16 Alo[16][32];

    const int t     = threadIdx.x;
    const int lane  = t & 31;
    const int wave  = t >> 5;
    const int m0    = blockIdx.x << 4;      // M tile base row
    const int nwave = wave << 6;            // 64 columns per wave

    // cooperative A-load coordinates: thread t loads float2 at (ar, ac)
    const int ar = t >> 4;
    const int ac = (t & 15) << 1;

    const int  mrow  = lane & 15;           // A row, B/C column within tile
    const bool hiLn  = (lane & 16) != 0;
    const int  abase = hiLn ? 8 : 0;        // A frag K base per ISA layout
    const int  bofs  = hiLn ? 8 : 0;        // B frag K base (lanes 16-31: K 8..15, 24..31)

    v8f accs[4] = {};

    for (int kt = 0; kt < F_IN / 32; ++kt) {
        const int kb = kt * 32;

        // ---- stage A tile (16x32 f32) as hi/lo f16 in LDS ----
        const float2 v = *(const float2*)(x + (size_t)(m0 + ar) * F_IN + kb + ac);
        const _Float16 hx = (_Float16)v.x;
        const _Float16 hy = (_Float16)v.y;
        Ahi[ar][ac]     = hx;  Alo[ar][ac]     = (_Float16)(v.x - (float)hx);
        Ahi[ar][ac + 1] = hy;  Alo[ar][ac + 1] = (_Float16)(v.y - (float)hy);
        __syncthreads();

        // ---- A fragments (16-bit A layout, ISA 7.12.2) ----
        v16h a_hi, a_lo;
        ((v8h*)&a_hi)[0] = *(const v8h*)&Ahi[mrow][abase];
        ((v8h*)&a_hi)[1] = *(const v8h*)&Ahi[mrow][abase + 16];
        ((v8h*)&a_lo)[0] = *(const v8h*)&Alo[mrow][abase];
        ((v8h*)&a_lo)[1] = *(const v8h*)&Alo[mrow][abase + 16];

        // ---- 4 N-tiles: load B frags, 3 WMMAs each (hi*hi + hi*lo + lo*hi) ----
        #pragma unroll
        for (int nt = 0; nt < 4; ++nt) {
            const size_t boff = (size_t)(nwave + nt * 16 + mrow) * F_IN + kb + bofs;
            v16h b_hi, b_lo;
            ((v8h*)&b_hi)[0] = *(const v8h*)(Bhi + boff);
            ((v8h*)&b_hi)[1] = *(const v8h*)(Bhi + boff + 16);
            ((v8h*)&b_lo)[0] = *(const v8h*)(Blo + boff);
            ((v8h*)&b_lo)[1] = *(const v8h*)(Blo + boff + 16);
            accs[nt] = __builtin_amdgcn_wmma_f32_16x16x32_f16(
                false, a_hi, false, b_hi, (short)0, accs[nt], false, false);
            accs[nt] = __builtin_amdgcn_wmma_f32_16x16x32_f16(
                false, a_hi, false, b_lo, (short)0, accs[nt], false, false);
            accs[nt] = __builtin_amdgcn_wmma_f32_16x16x32_f16(
                false, a_lo, false, b_hi, (short)0, accs[nt], false, false);
        }
        __syncthreads();
    }

    // ---- epilogue: +bias, split columns into xl / xr ----
    #pragma unroll
    for (int nt = 0; nt < 4; ++nt) {
        const int col = nwave + nt * 16 + mrow;
        float* dstb  = (col < H_) ? xl : xr;
        const int c  = (col < H_) ? col : col - H_;
        const float b = (col < H_) ? bl[c] : br[c];
        #pragma unroll
        for (int r = 0; r < 8; ++r) {
            const int row = m0 + r + (hiLn ? 8 : 0);   // C layout: lanes 16-31 -> M+8
            dstb[(size_t)row * H_ + c] = accs[nt][r] + b;
        }
    }
}

// ---------------------------------------------------------------------------
// Kernel 2: per-edge attention logits + segment max. One wave per edge.
// ---------------------------------------------------------------------------
__global__ __launch_bounds__(256)
void gat_edge_logits(const int* __restrict__ ei,
                     const float* __restrict__ xl, const float* __restrict__ xr,
                     const float* __restrict__ att,
                     float* __restrict__ e_arr, unsigned* __restrict__ emax)
{
    const int wid = blockIdx.x * 8 + (threadIdx.x >> 5);
    if (wid >= EN_) return;
    const int lane = threadIdx.x & 31;

    int src, dst;
    if (wid < E_EDGES) { src = ei[wid]; dst = ei[E_EDGES + wid]; }
    else               { src = dst = wid - E_EDGES; }          // self loop

    const int c0 = lane * 8;
    const float4 l0 = *(const float4*)(xl + (size_t)src * H_ + c0);
    const float4 l1 = *(const float4*)(xl + (size_t)src * H_ + c0 + 4);
    const float4 r0 = *(const float4*)(xr + (size_t)dst * H_ + c0);
    const float4 r1 = *(const float4*)(xr + (size_t)dst * H_ + c0 + 4);
    const float4 a0 = *(const float4*)(att + c0);
    const float4 a1 = *(const float4*)(att + c0 + 4);

    float s = 0.f, m;
    m = l0.x + r0.x; s += (m > 0.f ? m : 0.2f * m) * a0.x;
    m = l0.y + r0.y; s += (m > 0.f ? m : 0.2f * m) * a0.y;
    m = l0.z + r0.z; s += (m > 0.f ? m : 0.2f * m) * a0.z;
    m = l0.w + r0.w; s += (m > 0.f ? m : 0.2f * m) * a0.w;
    m = l1.x + r1.x; s += (m > 0.f ? m : 0.2f * m) * a1.x;
    m = l1.y + r1.y; s += (m > 0.f ? m : 0.2f * m) * a1.y;
    m = l1.z + r1.z; s += (m > 0.f ? m : 0.2f * m) * a1.z;
    m = l1.w + r1.w; s += (m > 0.f ? m : 0.2f * m) * a1.w;

    #pragma unroll
    for (int off = 16; off; off >>= 1) s += __shfl_xor(s, off, 32);

    if (lane == 0) {
        e_arr[wid] = s;
        atomicMax(emax + dst, encf(s));
    }
}

// ---------------------------------------------------------------------------
// Kernel 3: a = exp(e - emax[dst]) (in place) + segment denom. Thread/edge.
// ---------------------------------------------------------------------------
__global__ __launch_bounds__(256)
void gat_edge_exp(const int* __restrict__ ei, float* __restrict__ ea,
                  const unsigned* __restrict__ emax, float* __restrict__ denom)
{
    const int i = blockIdx.x * 256 + threadIdx.x;
    if (i >= EN_) return;
    const int dst = (i < E_EDGES) ? ei[E_EDGES + i] : (i - E_EDGES);
    const float a = __expf(ea[i] - decf(emax[dst]));
    ea[i] = a;
    atomicAdd(denom + dst, a);
}

// ---------------------------------------------------------------------------
// Kernel 4: out[dst] += (a/denom[dst]) * xl[src]. One wave per edge,
// 8 f32 atomics per lane; accumulator (32MB) is L2-resident.
// ---------------------------------------------------------------------------
__global__ __launch_bounds__(256)
void gat_edge_aggregate(const int* __restrict__ ei, const float* __restrict__ xl,
                        const float* __restrict__ a_arr,
                        const float* __restrict__ denom, float* __restrict__ outacc)
{
    const int wid = blockIdx.x * 8 + (threadIdx.x >> 5);
    if (wid >= EN_) return;
    const int lane = threadIdx.x & 31;

    int src, dst;
    if (wid < E_EDGES) { src = ei[wid]; dst = ei[E_EDGES + wid]; }
    else               { src = dst = wid - E_EDGES; }

    const float alpha = a_arr[wid] / denom[dst];
    const int c0 = lane * 8;
    const float4 l0 = *(const float4*)(xl + (size_t)src * H_ + c0);
    const float4 l1 = *(const float4*)(xl + (size_t)src * H_ + c0 + 4);
    float* o = outacc + (size_t)dst * H_ + c0;
    atomicAdd(o + 0, alpha * l0.x);
    atomicAdd(o + 1, alpha * l0.y);
    atomicAdd(o + 2, alpha * l0.z);
    atomicAdd(o + 3, alpha * l0.w);
    atomicAdd(o + 4, alpha * l1.x);
    atomicAdd(o + 5, alpha * l1.y);
    atomicAdd(o + 6, alpha * l1.z);
    atomicAdd(o + 7, alpha * l1.w);
}

// ---------------------------------------------------------------------------
// Kernel 5: per-graph prototype pooling + 2-layer MLP head. Block per graph.
// ---------------------------------------------------------------------------
__global__ __launch_bounds__(256)
void gat_pool_mlp(const float* __restrict__ outacc, const float* __restrict__ proto,
                  const float* __restrict__ bias,
                  const float* __restrict__ W1, const float* __restrict__ b1,
                  const float* __restrict__ W2, const float* __restrict__ b2,
                  float* __restrict__ out)
{
    __shared__ float p[NPG_];
    __shared__ float pooled[H_];
    __shared__ float y1[H_ / 2];
    const int g = blockIdx.x, t = threadIdx.x;

    for (int n = t; n < NPG_; n += 256) p[n] = proto[g * NPG_ + n];
    __syncthreads();

    const float bc = bias[t];
    const float* hb = outacc + (size_t)g * NPG_ * H_ + t;
    float s = 0.f;
    for (int n = 0; n < NPG_; ++n) {
        float h = hb[(size_t)n * H_] + bc;
        h = (h > 0.f) ? h : 0.001f * h;          // leaky_relu(0.001)
        s += p[n] * h;
    }
    pooled[t] = s;
    __syncthreads();

    if (t < H_ / 2) {
        float y = b1[t];
        for (int c = 0; c < H_; ++c) y += pooled[c] * W1[c * (H_ / 2) + t];
        y1[t] = (y > 0.f) ? y : 0.01f * y;       // leaky_relu(0.01)
    }
    __syncthreads();

    if (t < 2) {
        float y = b2[t];
        for (int j = 0; j < H_ / 2; ++j) y += y1[j] * W2[j * 2 + t];
        out[g * 2 + t] = y;
    }
}

// ---------------------------------------------------------------------------
extern "C" void kernel_launch(void* const* d_in, const int* in_sizes, int n_in,
                              void* d_out, int out_size, void* d_ws, size_t ws_size,
                              hipStream_t stream)
{
    const float* x     = (const float*)d_in[0];
    const int*   ei    = (const int*)  d_in[1];
    const float* proto = (const float*)d_in[2];
    const float* Wl    = (const float*)d_in[3];
    const float* bl    = (const float*)d_in[4];
    const float* Wr    = (const float*)d_in[5];
    const float* br    = (const float*)d_in[6];
    const float* att   = (const float*)d_in[7];
    const float* bias  = (const float*)d_in[8];
    const float* W1    = (const float*)d_in[9];
    const float* b1    = (const float*)d_in[10];
    const float* W2    = (const float*)d_in[11];
    const float* b2    = (const float*)d_in[12];
    float* out = (float*)d_out;

    // ---- carve workspace ----
    char* ws = (char*)d_ws;
    auto carve = [&](size_t bytes) -> char* {
        char* p = ws; ws += (bytes + 255) & ~(size_t)255; return p;
    };
    float*     xl     = (float*)    carve((size_t)N_NODES * H_ * 4);   // 32 MB
    float*     xr     = (float*)    carve((size_t)N_NODES * H_ * 4);   // 32 MB
    float*     outacc = (float*)    carve((size_t)N_NODES * H_ * 4);   // 32 MB
    float*     ea     = (float*)    carve((size_t)EN_ * 4);            // 4.3 MB
    unsigned*  emax   = (unsigned*) carve((size_t)N_NODES * 4);
    float*     denom  = (float*)    carve((size_t)N_NODES * 4);
    _Float16*  Bhi    = (_Float16*) carve((size_t)512 * F_IN * 2);
    _Float16*  Blo    = (_Float16*) carve((size_t)512 * F_IN * 2);

    // ---- zero accumulators (capturable async memsets) ----
    hipMemsetAsync(emax,   0, (size_t)N_NODES * 4, stream);       // enc identity
    hipMemsetAsync(denom,  0, (size_t)N_NODES * 4, stream);
    hipMemsetAsync(outacc, 0, (size_t)N_NODES * H_ * 4, stream);

    // ---- pipeline ----
    gat_split_w<<<(512 * F_IN + 255) / 256, 256, 0, stream>>>(Wl, Wr, Bhi, Blo);

    gat_gemm_xlxr<<<N_NODES / 16, 256, 0, stream>>>(x, Bhi, Blo, bl, br, xl, xr);

    gat_edge_logits<<<(EN_ + 7) / 8, 256, 0, stream>>>(ei, xl, xr, att, ea, emax);

    gat_edge_exp<<<(EN_ + 255) / 256, 256, 0, stream>>>(ei, ea, emax, denom);

    gat_edge_aggregate<<<(EN_ + 7) / 8, 256, 0, stream>>>(ei, xl, ea, denom, outacc);

    gat_pool_mlp<<<G_GR, 256, 0, stream>>>(outacc, proto, bias, W1, b1, W2, b2, out);
}
